// CBMPredictor_63780264345725
// MI455X (gfx1250) — matrix operand order repaired and checked by reading, not compiled, not run on hardware
//
#include <hip/hip_runtime.h>

#define NN 25600
#define EE 307200
#define BBATCH 256
#define HDIM 256
#define EDIM 16
#define KMIX 6
#define ZK 544          // padded K (528 -> 544, multiple of 32)
#define ZRAW 528
#define NLAYER 6
#define NODES_PER_GRAPH 100
#define ETILE 64        // edges per block in the WMMA message kernel

typedef __attribute__((ext_vector_type(16))) __bf16 v16bf;
typedef __attribute__((ext_vector_type(8)))  __bf16 v8bf;
typedef __attribute__((ext_vector_type(8)))  float  v8f;

__device__ __forceinline__ unsigned short f2b_bits(float f) {
  unsigned u = __builtin_bit_cast(unsigned, f);
  unsigned r = u + 0x7fffu + ((u >> 16) & 1u);     // round-nearest-even
  return (unsigned short)(r >> 16);
}
__device__ __forceinline__ float bf2f(unsigned lo16) {
  return __builtin_bit_cast(float, lo16 << 16);
}
__device__ __forceinline__ float silu_f(float x) { return x / (1.0f + __expf(-x)); }
__device__ __forceinline__ float ntn_f(float v, float pos, float neg) {
  if (__builtin_isnan(v)) return 0.0f;
  if (__builtin_isinf(v)) return v > 0.0f ? pos : neg;
  return v;
}

// ------------------------------------------------------------------ utilities
__global__ void zero_i32_kernel(int* __restrict__ p, int n) {
  int i = blockIdx.x * 256 + threadIdx.x; if (i < n) p[i] = 0;
}
__global__ void copy_i32_kernel(const int* __restrict__ in, int* __restrict__ out, int n) {
  int i = blockIdx.x * 256 + threadIdx.x; if (i < n) out[i] = in[i];
}
__global__ void f32_to_bf16_kernel(const float* __restrict__ in, unsigned short* __restrict__ out, int n) {
  int i = blockIdx.x * 256 + threadIdx.x; if (i < n) out[i] = f2b_bits(in[i]);
}
__global__ void ea_to_bf16_kernel(const float* __restrict__ ea, unsigned short* __restrict__ out, int n) {
  int i = blockIdx.x * 256 + threadIdx.x;
  if (i < n) out[i] = f2b_bits(ntn_f(ea[i], 1.0f, 0.0f));
}
// transpose+pad conv weights to bf16 [layer][n=256][k=544]
__global__ void prep_weights_kernel(const float* __restrict__ Wf, const float* __restrict__ Ws,
                                    unsigned short* __restrict__ WfT, unsigned short* __restrict__ WsT) {
  int i = blockIdx.x * 256 + threadIdx.x;
  if (i >= NLAYER * HDIM * ZK) return;
  int k = i % ZK; int n = (i / ZK) % HDIM; int l = i / (ZK * HDIM);
  unsigned short vf = 0, vs = 0;
  if (k < ZRAW) {
    int src = (l * ZRAW + k) * HDIM + n;
    vf = f2b_bits(Wf[src]); vs = f2b_bits(Ws[src]);
  }
  WfT[i] = vf; WsT[i] = vs;
}
// h0 = silu(ntn(x) @ gin_W + gin_b)
__global__ void encoder_kernel(const float* __restrict__ x, const float* __restrict__ W,
                               const float* __restrict__ b, float* __restrict__ h) {
  int i = blockIdx.x * 256 + threadIdx.x;
  if (i >= NN * HDIM) return;
  int n = i >> 8, c = i & 255;
  float acc = b[c];
#pragma unroll
  for (int k = 0; k < 7; k++) acc += ntn_f(x[n * 7 + k], 3.0f, -3.0f) * W[k * HDIM + c];
  h[i] = silu_f(acc);
}

// --------------------------------------------------------- CSR build (once)
__global__ void count_deg_kernel(const int* __restrict__ ei, int* __restrict__ cnt) {
  int i = blockIdx.x * 256 + threadIdx.x;
  if (i < EE) atomicAdd(cnt + ei[EE + i], 1);
}
// single-block exclusive scan over NN degree counts -> rowptr[NN+1]
__global__ void scan_kernel(const int* __restrict__ cnt, int* __restrict__ rowptr) {
  __shared__ int sdata[256];
  __shared__ int running;
  const int t = threadIdx.x;
  if (t == 0) { running = 0; rowptr[0] = 0; }
  __syncthreads();
  for (int chunk = 0; chunk < NN / 256; ++chunk) {
    int base = running;
    int v = cnt[chunk * 256 + t];
    sdata[t] = v; __syncthreads();
    for (int off = 1; off < 256; off <<= 1) {       // Hillis-Steele inclusive scan
      int add = (t >= off) ? sdata[t - off] : 0;
      __syncthreads();
      sdata[t] += add;
      __syncthreads();
    }
    rowptr[chunk * 256 + t + 1] = base + sdata[t];
    __syncthreads();
    if (t == 0) running = base + sdata[255];
    __syncthreads();
  }
}
__global__ void fill_perm_kernel(const int* __restrict__ ei, int* __restrict__ fillpos,
                                 int* __restrict__ perm) {
  int i = blockIdx.x * 256 + threadIdx.x;
  if (i >= EE) return;
  int pos = atomicAdd(fillpos + ei[EE + i], 1);
  perm[pos] = i;
}

// ------------------------------------------------- main WMMA edge-message GEMM
// 64 edges/block (CSR order via perm), 8 waves: wave = (mtile 0..3, nhalf 0..1).
// z-tile [64 x 544] bf16 in LDS (~68KB); messages written contiguously in CSR
// order as bf16 -- no float atomics anywhere.
__global__ __launch_bounds__(256) void edge_msg_wmma_kernel(
    const unsigned short* __restrict__ hbf,    // [NN,256] bf16
    const unsigned short* __restrict__ eabf,   // [EE,16] bf16
    const int* __restrict__ ei,                // [2,EE]
    const int* __restrict__ perm,              // [EE] CSR edge order
    const unsigned short* __restrict__ WfT,    // [256,544] bf16 (this layer)
    const unsigned short* __restrict__ WsT,
    const float* __restrict__ biasF,
    const float* __restrict__ biasS,
    unsigned short* __restrict__ msgbf)        // [EE,256] bf16, CSR row order
{
  __shared__ __align__(32) unsigned short zl[ETILE * ZK];
  const int e0 = blockIdx.x * ETILE;
  const int tid = threadIdx.x;

  { // gather: 4 threads per edge row
    const int row = tid >> 2, p = tid & 3;
    const int e = perm[e0 + row];
    const int s = ei[e];
    const int d = ei[EE + e];
    const uint4* gd = (const uint4*)(hbf + (size_t)d * HDIM + p * 64);
    const uint4* gs = (const uint4*)(hbf + (size_t)s * HDIM + p * 64);
    uint4* zd = (uint4*)(zl + row * ZK + p * 64);
    uint4* zs = (uint4*)(zl + row * ZK + 256 + p * 64);
#pragma unroll
    for (int i = 0; i < 8; i++) { zd[i] = gd[i]; zs[i] = gs[i]; }
    if (p == 0) {
      const uint4* ge = (const uint4*)(eabf + (size_t)e * EDIM);
      uint4* ze = (uint4*)(zl + row * ZK + 512);
      ze[0] = ge[0]; ze[1] = ge[1];
      uint4 zr; zr.x = zr.y = zr.z = zr.w = 0u;
      uint4* zp = (uint4*)(zl + row * ZK + ZRAW);
      zp[0] = zr; zp[1] = zr;                       // K padding 528..543
    }
  }
  __syncthreads();

  const int wid = tid >> 5, lane = tid & 31;
  const int mtile = wid >> 1;                 // 0..3
  const int nhalf = wid & 1;                  // 0..1
  const int arow  = (mtile << 4) + (lane & 15);
  const int ka    = (lane >> 4) << 4;         // A: K sub-offset 0 or 16
  const int ncol  = lane & 15;
  const int kb    = (lane >> 4) << 4;         // B: K sub-offset 0 or 16
  const int mbase = (mtile << 4) + ((lane >> 4) << 3);

  for (int nt = 0; nt < 8; ++nt) {
    const int n0 = ((nhalf << 3) + nt) << 4;  // 0..240
    v8f accF = {0, 0, 0, 0, 0, 0, 0, 0};
    v8f accS = {0, 0, 0, 0, 0, 0, 0, 0};
    for (int kc = 0; kc < 17; ++kc) {
      const unsigned short* ap = zl + arow * ZK + kc * 32 + ka;
      v8bf lo = *(const v8bf*)(ap);
      v8bf hi = *(const v8bf*)(ap + 8);
      v16bf A;
#pragma unroll
      for (int i = 0; i < 8; i++) { A[i] = lo[i]; A[i + 8] = hi[i]; }
      const int bo = (n0 + ncol) * ZK + kc * 32 + kb;
      v16bf Bf = *(const v16bf*)(WfT + bo);
      v16bf Bs = *(const v16bf*)(WsT + bo);
      accF = __builtin_amdgcn_wmma_f32_16x16x32_bf16(false, A, false, Bf, (short)0, accF, false, false);
      accS = __builtin_amdgcn_wmma_f32_16x16x32_bf16(false, A, false, Bs, (short)0, accS, false, false);
    }
    const int n = n0 + ncol;
    const float bF = biasF[n], bS = biasS[n];
#pragma unroll
    for (int r = 0; r < 8; r++) {
      float f = accF[r] + bF;
      float s = accS[r] + bS;
      float gate = 1.0f / (1.0f + __expf(-f));
      float sp = (s > 20.0f) ? s : __logf(1.0f + __expf(s));
      msgbf[(size_t)(e0 + mbase + r) * HDIM + n] = f2b_bits(gate * sp);
    }
  }
}

// fused CSR aggregation + residual + LN: h' = LN(silu(h + sum(msg)) + h)
// wave-per-node (wave32); msgbuf rows for a node are contiguous (CSR order)
__global__ __launch_bounds__(256) void node_update_kernel(
    const float* __restrict__ hin, const unsigned short* __restrict__ msgbf,
    const int* __restrict__ rowptr,
    const float* __restrict__ g, const float* __restrict__ bb,
    float* __restrict__ hout)
{
  const int wid = threadIdx.x >> 5, lane = threadIdx.x & 31;
  const int node = (blockIdx.x << 3) + wid;
  const size_t base = (size_t)node * HDIM + lane * 8;
  const int j0 = rowptr[node], j1 = rowptr[node + 1];
  float acc[8] = {0, 0, 0, 0, 0, 0, 0, 0};
  for (int j = j0; j < j1; ++j) {
    const uint4 mv = *(const uint4*)(msgbf + (size_t)j * HDIM + lane * 8);
    acc[0] += bf2f(mv.x & 0xffffu); acc[1] += bf2f(mv.x >> 16);
    acc[2] += bf2f(mv.y & 0xffffu); acc[3] += bf2f(mv.y >> 16);
    acc[4] += bf2f(mv.z & 0xffffu); acc[5] += bf2f(mv.z >> 16);
    acc[6] += bf2f(mv.w & 0xffffu); acc[7] += bf2f(mv.w >> 16);
  }
  float xv[8]; float s = 0.0f, q = 0.0f;
#pragma unroll
  for (int i = 0; i < 8; i++) {
    float hv = hin[base + i];
    float xi = silu_f(hv + acc[i]) + hv;
    xv[i] = xi; s += xi; q += xi * xi;
  }
#pragma unroll
  for (int m = 16; m > 0; m >>= 1) { s += __shfl_xor(s, m, 32); q += __shfl_xor(q, m, 32); }
  const float mean = s * (1.0f / HDIM);
  const float var  = q * (1.0f / HDIM) - mean * mean;
  const float inv  = rsqrtf(var + 1e-5f);
#pragma unroll
  for (int i = 0; i < 8; i++) {
    int c = lane * 8 + i;
    hout[base + i] = (xv[i] - mean) * inv * g[c] + bb[c];
  }
}

__global__ void pool_kernel(const float* __restrict__ h, float* __restrict__ pooled) {
  int b = blockIdx.x, c = threadIdx.x;
  const float* hp = h + (size_t)b * NODES_PER_GRAPH * HDIM + c;
  float s = 0.0f;
  for (int i = 0; i < NODES_PER_GRAPH; i++) s += hp[(size_t)i * HDIM];
  pooled[b * HDIM + c] = s * (1.0f / NODES_PER_GRAPH);
}

// generic out = act(in @ W + bias); W is [K,M]
__global__ void dense_kernel(const float* __restrict__ in, const float* __restrict__ W,
                             const float* __restrict__ bias, float* __restrict__ out,
                             int K, int M, int total, int act, int ntn3) {
  int i = blockIdx.x * 256 + threadIdx.x;
  if (i >= total) return;
  int b = i / M, j = i % M;
  float acc = bias[j];
  const float* ir = in + (size_t)b * K;
  for (int k = 0; k < K; k++) {
    float v = ir[k];
    if (ntn3) v = ntn_f(v, 3.0f, -3.0f);
    acc += v * W[(size_t)k * M + j];
  }
  if (act) acc = silu_f(acc);
  out[i] = acc;
}

__global__ void ln_kernel(float* __restrict__ x, const float* __restrict__ g,
                          const float* __restrict__ b, int M) {
  __shared__ float s1[128]; __shared__ float s2[128];
  const int row = blockIdx.x, t = threadIdx.x;
  float* xr = x + (size_t)row * M;
  float a = 0.0f, q = 0.0f;
  for (int c = t; c < M; c += 128) { float v = xr[c]; a += v; q += v * v; }
  s1[t] = a; s2[t] = q; __syncthreads();
  for (int st = 64; st > 0; st >>= 1) {
    if (t < st) { s1[t] += s1[t + st]; s2[t] += s2[t + st]; }
    __syncthreads();
  }
  float mean = s1[0] / M;
  float var  = s2[0] / M - mean * mean;
  float inv  = rsqrtf(var + 1e-5f);
  for (int c = t; c < M; c += 128) { float v = xr[c]; xr[c] = (v - mean) * inv * g[c] + b[c]; }
}

__global__ void concat_kernel(const float* __restrict__ zg, const float* __restrict__ zt,
                              const float* __restrict__ zm, float* __restrict__ zf) {
  int i = blockIdx.x * 256 + threadIdx.x;
  if (i >= BBATCH * 384) return;
  int b = i / 384, j = i % 384;
  float v;
  if (j < 192) v = zg[b * 192 + j];
  else if (j < 320) v = zt[b * 128 + (j - 192)];
  else v = zm[b * 64 + (j - 320)];
  if (__builtin_isnan(v)) v = 0.0f;
  else if (__builtin_isinf(v)) v = v > 0.0f ? 3.4e38f : -3.4e38f;
  zf[i] = v;
}

__global__ void softmax6_kernel(const float* __restrict__ in, float* __restrict__ out) {
  int b = threadIdx.x;
  float mx = -1e30f;
  for (int k = 0; k < KMIX; k++) mx = fmaxf(mx, in[b * KMIX + k]);
  float e[KMIX]; float s = 0.0f;
  for (int k = 0; k < KMIX; k++) { e[k] = __expf(in[b * KMIX + k] - mx); s += e[k]; }
  float inv = 1.0f / s;
  for (int k = 0; k < KMIX; k++) out[b * KMIX + k] = e[k] * inv;
}

// per-(b,k) poly branch: lin + q + c
__global__ __launch_bounds__(192) void poly_kernel(
    const float* __restrict__ z,
    const float* __restrict__ plinW, const float* __restrict__ plinb,
    const float* __restrict__ pq1W, const float* __restrict__ pq1b,
    const float* __restrict__ pq2W, const float* __restrict__ pq2b,
    const float* __restrict__ pc1W, const float* __restrict__ pc1b,
    const float* __restrict__ pc2W, const float* __restrict__ pc2b,
    float* __restrict__ sbk)
{
  __shared__ float red[192];
  const int b = blockIdx.x, k = blockIdx.y, m = threadIdx.x;
  const float* zr = z + (size_t)b * 384;
  float acc = pq1b[k * 192 + m];
  const float* w = pq1W + (size_t)k * 384 * 192 + m;
  for (int d = 0; d < 384; ++d) acc += zr[d] * w[(size_t)d * 192];
  red[m] = silu_f(acc) * pq2W[k * 192 + m];
  __syncthreads();
  for (int st = 128; st > 0; st >>= 1) { if (m < st && m + st < 192) red[m] += red[m + st]; __syncthreads(); }
  float qv = red[0] + pq2b[k];
  __syncthreads();
  float cterm = 0.0f;
  if (m < 96) {
    float a2 = pc1b[k * 96 + m];
    const float* w2 = pc1W + (size_t)k * 384 * 96 + m;
    for (int d = 0; d < 384; ++d) a2 += zr[d] * w2[(size_t)d * 96];
    float hc = silu_f(a2); hc = hc * hc;
    cterm = hc * pc2W[k * 96 + m];
  }
  red[m] = cterm;
  __syncthreads();
  for (int st = 128; st > 0; st >>= 1) { if (m < st && m + st < 192) red[m] += red[m + st]; __syncthreads(); }
  float cv = red[0] + pc2b[k];
  __syncthreads();
  red[m] = zr[m] * plinW[k * 384 + m] + zr[m + 192] * plinW[k * 384 + m + 192];
  __syncthreads();
  for (int st = 128; st > 0; st >>= 1) { if (m < st && m + st < 192) red[m] += red[m + st]; __syncthreads(); }
  if (m == 0) sbk[b * KMIX + k] = (red[0] + plinb[k]) + qv + cv;
}

__global__ void final_kernel(const float* __restrict__ gates, const float* __restrict__ sbk,
                             float* __restrict__ y) {
  int b = threadIdx.x;
  float acc = 0.0f;
  for (int k = 0; k < KMIX; k++) acc += gates[b * KMIX + k] * sbk[b * KMIX + k];
  y[b] = acc;
}

// ---------------------------------------------------------------------- host
extern "C" void kernel_launch(void* const* d_in, const int* in_sizes, int n_in,
                              void* d_out, int out_size, void* d_ws, size_t ws_size,
                              hipStream_t stream) {
  (void)in_sizes; (void)n_in; (void)out_size; (void)ws_size;
  const float* x         = (const float*)d_in[0];
  const float* edge_attr = (const float*)d_in[1];
  const float* tda       = (const float*)d_in[2];
  const float* megnet    = (const float*)d_in[3];
  const int*   ei        = (const int*)d_in[4];
  // d_in[5] = batch (structure known: 100 contiguous nodes/graph)
  const float* gin_W = (const float*)d_in[6];  const float* gin_b = (const float*)d_in[7];
  const float* cWf   = (const float*)d_in[8];  const float* cbf   = (const float*)d_in[9];
  const float* cWs   = (const float*)d_in[10]; const float* cbs   = (const float*)d_in[11];
  const float* gln_g = (const float*)d_in[12]; const float* gln_b = (const float*)d_in[13];
  const float* gout_W = (const float*)d_in[14]; const float* gout_b = (const float*)d_in[15];
  const float* t1_W = (const float*)d_in[16]; const float* t1_b = (const float*)d_in[17];
  const float* tln1_g = (const float*)d_in[18]; const float* tln1_b = (const float*)d_in[19];
  const float* t2_W = (const float*)d_in[20]; const float* t2_b = (const float*)d_in[21];
  const float* tln2_g = (const float*)d_in[22]; const float* tln2_b = (const float*)d_in[23];
  const float* t3_W = (const float*)d_in[24]; const float* t3_b = (const float*)d_in[25];
  const float* m1_W = (const float*)d_in[26]; const float* m1_b = (const float*)d_in[27];
  const float* mln1_g = (const float*)d_in[28]; const float* mln1_b = (const float*)d_in[29];
  const float* m2_W = (const float*)d_in[30]; const float* m2_b = (const float*)d_in[31];
  const float* mln2_g = (const float*)d_in[32]; const float* mln2_b = (const float*)d_in[33];
  const float* m3_W = (const float*)d_in[34]; const float* m3_b = (const float*)d_in[35];
  const float* fln_g = (const float*)d_in[36]; const float* fln_b = (const float*)d_in[37];
  const float* g1_W = (const float*)d_in[38]; const float* g1_b = (const float*)d_in[39];
  const float* hln_g = (const float*)d_in[40]; const float* hln_b = (const float*)d_in[41];
  const float* g2_W = (const float*)d_in[42]; const float* g2_b = (const float*)d_in[43];
  const float* g3_W = (const float*)d_in[44]; const float* g3_b = (const float*)d_in[45];
  const float* plin_W = (const float*)d_in[46]; const float* plin_b = (const float*)d_in[47];
  const float* pq1_W = (const float*)d_in[48]; const float* pq1_b = (const float*)d_in[49];
  const float* pq2_W = (const float*)d_in[50]; const float* pq2_b = (const float*)d_in[51];
  const float* pc1_W = (const float*)d_in[52]; const float* pc1_b = (const float*)d_in[53];
  const float* pc2_W = (const float*)d_in[54]; const float* pc2_b = (const float*)d_in[55];

  char* wp = (char*)d_ws;
  auto alloc = [&](size_t bytes) { char* p = wp; wp += (bytes + 255) & ~(size_t)255; return p; };
  float* hA   = (float*)alloc((size_t)NN * HDIM * 4);
  unsigned short* hbf   = (unsigned short*)alloc((size_t)NN * HDIM * 2);
  unsigned short* eabf  = (unsigned short*)alloc((size_t)EE * EDIM * 2);
  unsigned short* WfT   = (unsigned short*)alloc((size_t)NLAYER * HDIM * ZK * 2);
  unsigned short* WsT   = (unsigned short*)alloc((size_t)NLAYER * HDIM * ZK * 2);
  unsigned short* msgbf = (unsigned short*)alloc((size_t)EE * HDIM * 2);   // 157 MB
  int* cnt     = (int*)alloc((size_t)NN * 4);
  int* rowptr  = (int*)alloc((size_t)(NN + 1) * 4);
  int* fillpos = (int*)alloc((size_t)NN * 4);
  int* perm    = (int*)alloc((size_t)EE * 4);
  float* pooled = (float*)alloc((size_t)BBATCH * 256 * 4);
  float* buf256 = (float*)alloc((size_t)BBATCH * 256 * 4);
  float* zgnn   = (float*)alloc((size_t)BBATCH * 192 * 4);
  float* ztda   = (float*)alloc((size_t)BBATCH * 128 * 4);
  float* buf128 = (float*)alloc((size_t)BBATCH * 128 * 4);
  float* zmeg   = (float*)alloc((size_t)BBATCH * 64 * 4);
  float* g1buf  = (float*)alloc((size_t)BBATCH * 48 * 4);
  float* g2buf  = (float*)alloc((size_t)BBATCH * 24 * 4);
  float* g3buf  = (float*)alloc((size_t)BBATCH * KMIX * 4);
  float* gates  = (float*)alloc((size_t)BBATCH * KMIX * 4);
  float* sbk    = (float*)alloc((size_t)BBATCH * KMIX * 4);

  float* yout = (float*)d_out;              // [B]
  float* zf   = (float*)d_out + BBATCH;     // [B,384]

  auto gr = [](int total) { return dim3((unsigned)((total + 255) / 256)); };

  // --- one-time prep: weights, edge attrs, CSR permutation of edges by dst
  prep_weights_kernel<<<gr(NLAYER * HDIM * ZK), 256, 0, stream>>>(cWf, cWs, WfT, WsT);
  ea_to_bf16_kernel<<<gr(EE * EDIM), 256, 0, stream>>>(edge_attr, eabf, EE * EDIM);
  zero_i32_kernel<<<gr(NN), 256, 0, stream>>>(cnt, NN);
  count_deg_kernel<<<gr(EE), 256, 0, stream>>>(ei, cnt);
  scan_kernel<<<1, 256, 0, stream>>>(cnt, rowptr);
  copy_i32_kernel<<<gr(NN), 256, 0, stream>>>(rowptr, fillpos, NN);
  fill_perm_kernel<<<gr(EE), 256, 0, stream>>>(ei, fillpos, perm);

  encoder_kernel<<<gr(NN * HDIM), 256, 0, stream>>>(x, gin_W, gin_b, hA);

  // --- 6 CGConv layers: WMMA message GEMM (CSR order) + fused CSR-reduce + LN
  for (int l = 0; l < NLAYER; ++l) {
    f32_to_bf16_kernel<<<gr(NN * HDIM), 256, 0, stream>>>(hA, hbf, NN * HDIM);
    edge_msg_wmma_kernel<<<EE / ETILE, 256, 0, stream>>>(
        hbf, eabf, ei, perm,
        WfT + (size_t)l * HDIM * ZK, WsT + (size_t)l * HDIM * ZK,
        cbf + l * HDIM, cbs + l * HDIM, msgbf);
    node_update_kernel<<<NN / 8, 256, 0, stream>>>(
        hA, msgbf, rowptr, gln_g + l * HDIM, gln_b + l * HDIM, hA);
  }

  // --- pooling + GNN head
  pool_kernel<<<BBATCH, 256, 0, stream>>>(hA, pooled);
  dense_kernel<<<gr(BBATCH * 192), 256, 0, stream>>>(pooled, gout_W, gout_b, zgnn, 256, 192, BBATCH * 192, 0, 0);

  // --- TDA projector 32->256->256->128
  dense_kernel<<<gr(BBATCH * 256), 256, 0, stream>>>(tda, t1_W, t1_b, buf256, 32, 256, BBATCH * 256, 1, 1);
  ln_kernel<<<BBATCH, 128, 0, stream>>>(buf256, tln1_g, tln1_b, 256);
  dense_kernel<<<gr(BBATCH * 256), 256, 0, stream>>>(buf256, t2_W, t2_b, pooled, 256, 256, BBATCH * 256, 1, 0);
  ln_kernel<<<BBATCH, 128, 0, stream>>>(pooled, tln2_g, tln2_b, 256);
  dense_kernel<<<gr(BBATCH * 128), 256, 0, stream>>>(pooled, t3_W, t3_b, ztda, 256, 128, BBATCH * 128, 0, 0);

  // --- MEGNet projector 16->128->128->64
  dense_kernel<<<gr(BBATCH * 128), 256, 0, stream>>>(megnet, m1_W, m1_b, buf128, 16, 128, BBATCH * 128, 1, 1);
  ln_kernel<<<BBATCH, 128, 0, stream>>>(buf128, mln1_g, mln1_b, 128);
  dense_kernel<<<gr(BBATCH * 128), 256, 0, stream>>>(buf128, m2_W, m2_b, buf256, 128, 128, BBATCH * 128, 1, 0);
  ln_kernel<<<BBATCH, 128, 0, stream>>>(buf256, mln2_g, mln2_b, 128);
  dense_kernel<<<gr(BBATCH * 64), 256, 0, stream>>>(buf256, m3_W, m3_b, zmeg, 128, 64, BBATCH * 64, 0, 0);

  // --- fusion: z = LN(concat) -> written straight into d_out[B:]
  concat_kernel<<<gr(BBATCH * 384), 256, 0, stream>>>(zgnn, ztda, zmeg, zf);
  ln_kernel<<<BBATCH, 128, 0, stream>>>(zf, fln_g, fln_b, 384);

  // --- CPPN gate 384->48->24->6 + softmax
  dense_kernel<<<gr(BBATCH * 48), 256, 0, stream>>>(zf, g1_W, g1_b, g1buf, 384, 48, BBATCH * 48, 1, 0);
  ln_kernel<<<BBATCH, 128, 0, stream>>>(g1buf, hln_g, hln_b, 48);
  dense_kernel<<<gr(BBATCH * 24), 256, 0, stream>>>(g1buf, g2_W, g2_b, g2buf, 48, 24, BBATCH * 24, 1, 0);
  dense_kernel<<<gr(BBATCH * KMIX), 256, 0, stream>>>(g2buf, g3_W, g3_b, g3buf, 24, KMIX, BBATCH * KMIX, 0, 0);
  softmax6_kernel<<<1, BBATCH, 0, stream>>>(g3buf, gates);

  // --- poly branches + mixture
  poly_kernel<<<dim3(BBATCH, KMIX), 192, 0, stream>>>(
      zf, plin_W, plin_b, pq1_W, pq1_b, pq2_W, pq2_b, pc1_W, pc1_b, pc2_W, pc2_b, sbk);
  final_kernel<<<1, BBATCH, 0, stream>>>(gates, sbk, yout);
}
